// EBP_binaryNet_88862873354480
// MI455X (gfx1250) — compile-verified, matrix operand order checked by reading, not run on hardware
//
#include <hip/hip_runtime.h>

// ---------------------------------------------------------------------------
// EBP binary-net forward on MI455X (gfx1250): bf16 WMMA GEMMs + f32 epilogues
// Register-blocked: each wave computes a 64x32 output block (8 WMMA accums).
// ---------------------------------------------------------------------------

typedef __attribute__((ext_vector_type(16))) __bf16 v16bf;
typedef __attribute__((ext_vector_type(8)))  __bf16 v8bf;
typedef __attribute__((ext_vector_type(8)))  float  v8f;

#define SQ2PI 0.79788456f
#define BN_EPS 1e-5f

constexpr int M_ROWS = 65536;
constexpr int IN_DIM = 784;
constexpr int KP0    = 800;   // 784 padded to a multiple of 32
constexpr int H_DIM  = 512;
constexpr int N_OUT  = 10;

constexpr int MT = 4;         // 16-row subtiles per wave  (64 rows)
constexpr int NT = 2;         // 16-col subtiles per wave  (32 cols)

// ---- workspace layout (bytes; d_ws is 256-aligned from hipMalloc) ----------
constexpr size_t align_up(size_t v) { return (v + 255) & ~size_t(255); }

constexpr size_t SMALL_FLOATS = 512 * 9 + 2;               // diags, col stats, accums
constexpr size_t OFF_SMALL = 0;
constexpr size_t OFF_XBF   = OFF_SMALL + align_up(SMALL_FLOATS * 4);
constexpr size_t OFF_M0T   = OFF_XBF   + align_up((size_t)M_ROWS * KP0 * 2);
constexpr size_t OFF_M1T   = OFF_M0T   + align_up((size_t)H_DIM * KP0 * 2);
constexpr size_t OFF_M3T   = OFF_M1T   + align_up((size_t)H_DIM * H_DIM * 2);
constexpr size_t OFF_MLT   = OFF_M3T   + align_up((size_t)H_DIM * H_DIM * 2);
constexpr size_t OFF_H     = OFF_MLT   + align_up((size_t)16 * H_DIM * 2);
constexpr size_t OFF_ACTA  = OFF_H     + align_up((size_t)M_ROWS * H_DIM * 4);
constexpr size_t OFF_ACTB  = OFF_ACTA  + align_up((size_t)M_ROWS * H_DIM * 2);

// ---------------------------------------------------------------------------
__global__ void k_zero_f32(float* p, int n) {
    int i = blockIdx.x * blockDim.x + threadIdx.x;
    if (i < n) p[i] = 0.0f;
}

// x[M,784] f32 -> xbf[M,800] bf16 (zero padded)
__global__ void k_convert_x(const float* __restrict__ x, __bf16* __restrict__ xb) {
    long long e = (long long)blockIdx.x * blockDim.x + threadIdx.x;
    if (e >= (long long)M_ROWS * KP0) return;
    int row = (int)(e / KP0), k = (int)(e % KP0);
    float v = (k < IN_DIM) ? x[(long long)row * IN_DIM + k] : 0.0f;
    xb[e] = (__bf16)v;
}

// m0 = tanh(w0/2); store transposed [512 x 800] bf16; diag1[j] = sum_i (1-m^2)
__global__ void k_prep_m0(const float* __restrict__ w0, __bf16* __restrict__ m0T,
                          float* __restrict__ diag1) {
    int e = blockIdx.x * blockDim.x + threadIdx.x;
    if (e >= H_DIM * KP0) return;
    int j = e / KP0, i = e % KP0;
    float m = 0.0f;
    if (i < IN_DIM) {
        m = tanhf(0.5f * w0[i * H_DIM + j]);
        atomicAdd(diag1 + j, 1.0f - m * m);
    }
    m0T[j * KP0 + i] = (__bf16)m;
}

// square weight: m = tanh(w/2); store transposed [512 x 512]; diag[r] = sum over row r
__global__ void k_prep_sq(const float* __restrict__ w, __bf16* __restrict__ mT,
                          float* __restrict__ diag) {
    int e = blockIdx.x * blockDim.x + threadIdx.x;
    if (e >= H_DIM * H_DIM) return;
    int r = e / H_DIM, c = e % H_DIM;
    float m = tanhf(0.5f * w[r * H_DIM + c]);
    mT[c * H_DIM + r] = (__bf16)m;
    atomicAdd(diag + r, 1.0f - m * m);   // axis=1 sum (faithful to reference)
}

// mlast [512 x 10] -> transposed, N padded to 16: mT[n*512 + k]
__global__ void k_prep_mlast(const float* __restrict__ wlast, __bf16* __restrict__ mT) {
    int e = blockIdx.x * blockDim.x + threadIdx.x;
    if (e >= 16 * H_DIM) return;
    int n = e / H_DIM, k = e % H_DIM;
    float m = (n < N_OUT) ? tanhf(0.5f * wlast[k * N_OUT + n]) : 0.0f;
    mT[n * H_DIM + k] = (__bf16)m;
}

// ---------------------------------------------------------------------------
// 16-bit A-matrix 16x32 lane layout (ISA 7.12.2):
//   lanes 0-15 : row M=l, v0..v3 = K {0..7},  v4..v7 = K {16..23}
//   lanes16-31 : row M=l, v0..v3 = K {8..15}, v4..v7 = K {24..31}
// Both A and B^T are row-major along K, so each lane does two 16B loads.
union FragBF { v16bf v; v8bf h[2]; };

__device__ __forceinline__ v16bf load_frag(const __bf16* __restrict__ base,
                                           int ld, int row, int k0, int half) {
    FragBF f;
    const __bf16* p = base + (size_t)row * ld + k0 + half * 8;
    f.h[0] = *(const v8bf*)p;          // K = half*8 + 0..7
    f.h[1] = *(const v8bf*)(p + 16);   // K = 16 + half*8 + 0..7
    return f.v;
}

// C = A[M,K] * B^T[N,K]^T ; EBP epilogue: h = SQ2PI*(c+th[n])*rsqrt(diag[n]),
// store fp32 h and accumulate per-column sum / sumsq for BatchNorm.
// One wave computes MT*16 x NT*16 = 64x32 of C (8 WMMA accumulators).
__global__ void k_gemm_ebp(const __bf16* __restrict__ A, const __bf16* __restrict__ BT,
                           const float* __restrict__ th, const float* __restrict__ diag,
                           float* __restrict__ Hout, float* __restrict__ csum,
                           float* __restrict__ csumsq, int N, int K, int lda) {
    int wave = blockIdx.x * (blockDim.x >> 5) + (threadIdx.x >> 5);
    int ntn = N / (NT * 16);
    int ntm = M_ROWS / (MT * 16);
    if (wave >= ntm * ntn) return;               // wave-uniform: EXEC all-1 for WMMA
    int tm = wave / ntn, tn = wave % ntn;        // tn fastest: waves share A block
    int lane = threadIdx.x & 31;
    int half = lane >> 4, l = lane & 15;
    int am0 = tm * (MT * 16) + l;
    int bn0 = tn * (NT * 16) + l;

    v8f acc[MT][NT] = {};
#pragma unroll 2
    for (int k0 = 0; k0 < K; k0 += 32) {
        v16bf b[NT], a[MT];
#pragma unroll
        for (int ni = 0; ni < NT; ++ni)
            b[ni] = load_frag(BT, K, bn0 + ni * 16, k0, half);
#pragma unroll
        for (int mi = 0; mi < MT; ++mi)
            a[mi] = load_frag(A, lda, am0 + mi * 16, k0, half);
#pragma unroll
        for (int mi = 0; mi < MT; ++mi)
#pragma unroll
            for (int ni = 0; ni < NT; ++ni)
                acc[mi][ni] = __builtin_amdgcn_wmma_f32_16x16x32_bf16(
                    false, a[mi], false, b[ni], (short)0, acc[mi][ni], false, false);
    }

#pragma unroll
    for (int ni = 0; ni < NT; ++ni) {
        int n = tn * (NT * 16) + ni * 16 + l;    // lanes l and l+16 share column n
        float scale = SQ2PI * rsqrtf(diag[n]);
        float bias  = th[n];
        float s = 0.0f, sq = 0.0f;
#pragma unroll
        for (int mi = 0; mi < MT; ++mi) {
            int mbase = tm * (MT * 16) + mi * 16 + half * 8;  // vgpr r -> row half*8+r
#pragma unroll
            for (int r = 0; r < 8; ++r) {
                float hv = scale * (acc[mi][ni][r] + bias);
                Hout[(size_t)(mbase + r) * N + n] = hv;
                s  += hv;
                sq += hv * hv;
            }
        }
        s  += __shfl_down(s, 16, 32);
        sq += __shfl_down(sq, 16, 32);
        if (half == 0) {
            atomicAdd(csum + n, s);
            atomicAdd(csumsq + n, sq);
        }
    }
}

// BatchNorm (training-mode, biased var) + tanh, emit bf16 for next GEMM
__global__ void k_bn_tanh(const float* __restrict__ H, const float* __restrict__ csum,
                          const float* __restrict__ csumsq, __bf16* __restrict__ Xout) {
    long long idx = (long long)blockIdx.x * blockDim.x + threadIdx.x;
    if (idx >= (long long)M_ROWS * H_DIM) return;
    int j = (int)(idx & (H_DIM - 1));
    const float invM = 1.0f / (float)M_ROWS;
    float mu  = csum[j] * invM;
    float var = csumsq[j] * invM - mu * mu;
    float xn  = (H[idx] - mu) * rsqrtf(var + BN_EPS);
    Xout[idx] = (__bf16)tanhf(xn);
}

// head GEMM: N padded to 16, keep n<10; hlastbar -> d_out[0 : M*10]
// One wave computes 64x16 (MT=4, single N tile; B frag reused 4x).
__global__ void k_gemm_final(const __bf16* __restrict__ A, const __bf16* __restrict__ BT,
                             const float* __restrict__ thlast, float* __restrict__ out) {
    int wave = blockIdx.x * (blockDim.x >> 5) + (threadIdx.x >> 5);
    if (wave >= M_ROWS / (MT * 16)) return;
    int tm = wave;
    int lane = threadIdx.x & 31;
    int half = lane >> 4, l = lane & 15;
    int am0 = tm * (MT * 16) + l;

    v8f acc[MT] = {};
#pragma unroll 2
    for (int k0 = 0; k0 < H_DIM; k0 += 32) {
        v16bf b = load_frag(BT, H_DIM, l, k0, half);
#pragma unroll
        for (int mi = 0; mi < MT; ++mi) {
            v16bf a = load_frag(A, H_DIM, am0 + mi * 16, k0, half);
            acc[mi] = __builtin_amdgcn_wmma_f32_16x16x32_bf16(
                false, a, false, b, (short)0, acc[mi], false, false);
        }
    }
    if (l < N_OUT) {
        float bias = thlast[l];
#pragma unroll
        for (int mi = 0; mi < MT; ++mi) {
            int mbase = tm * (MT * 16) + mi * 16 + half * 8;
#pragma unroll
            for (int r = 0; r < 8; ++r)
                out[(size_t)(mbase + r) * N_OUT + l] = acc[mi][r] + bias;
        }
    }
}

// per-row log-softmax, NLL loss sum and correct count (block-reduced atomics)
__global__ void k_softmax_loss(const float* __restrict__ hlast, const int* __restrict__ target,
                               float* __restrict__ logp, float* __restrict__ accums) {
    __shared__ float s_loss, s_corr;
    if (threadIdx.x == 0) { s_loss = 0.0f; s_corr = 0.0f; }
    __syncthreads();

    int row = blockIdx.x * blockDim.x + threadIdx.x;
    float loss = 0.0f, corr = 0.0f;
    if (row < M_ROWS) {
        float v[N_OUT];
        float mx = -3.0e38f;
        int am = 0;
        for (int i = 0; i < N_OUT; ++i) {
            v[i] = hlast[(size_t)row * N_OUT + i];
            if (v[i] > mx) { mx = v[i]; am = i; }
        }
        float se = 0.0f;
        for (int i = 0; i < N_OUT; ++i) se += expf(v[i] - mx);
        float lz = mx + logf(se);
        for (int i = 0; i < N_OUT; ++i) logp[(size_t)row * N_OUT + i] = v[i] - lz;
        int t = target[row];
        loss = -(v[t] - lz);
        corr = (am == t) ? 1.0f : 0.0f;
    }
    atomicAdd(&s_loss, loss);
    atomicAdd(&s_corr, corr);
    __syncthreads();
    if (threadIdx.x == 0) {
        atomicAdd(accums + 0, s_loss);
        atomicAdd(accums + 1, s_corr);
    }
}

__global__ void k_finalize(const float* __restrict__ accums, float* __restrict__ out2) {
    const float invM = 1.0f / (float)M_ROWS;
    out2[0] = accums[0] * invM;   // expected_loss
    out2[1] = accums[1] * invM;   // fraction_correct
}

// ---------------------------------------------------------------------------
extern "C" void kernel_launch(void* const* d_in, const int* in_sizes, int n_in,
                              void* d_out, int out_size, void* d_ws, size_t ws_size,
                              hipStream_t stream) {
    (void)in_sizes; (void)n_in; (void)out_size; (void)ws_size;

    const float* x      = (const float*)d_in[0];
    const int*   target = (const int*)  d_in[1];
    const float* w0     = (const float*)d_in[2];
    const float* w1     = (const float*)d_in[3];
    /* w2 = d_in[4] unused (faithful to reference) */
    const float* w3     = (const float*)d_in[5];
    const float* wlast  = (const float*)d_in[6];
    const float* th0    = (const float*)d_in[7];
    const float* th1    = (const float*)d_in[8];
    /* th2 = d_in[9] unused */
    const float* thlast = (const float*)d_in[10];
    float* out = (float*)d_out;

    char* ws = (char*)d_ws;
    float*  sm      = (float*)(ws + OFF_SMALL);
    float*  diag1   = sm + 0,    *diag2 = sm + 512,  *diag3 = sm + 1024;
    float*  csum0   = sm + 1536, *csq0  = sm + 2048;
    float*  csum1   = sm + 2560, *csq1  = sm + 3072;
    float*  csum2   = sm + 3584, *csq2  = sm + 4096;
    float*  accums  = sm + 4608;
    __bf16* xbf     = (__bf16*)(ws + OFF_XBF);
    __bf16* m0T     = (__bf16*)(ws + OFF_M0T);
    __bf16* m1T     = (__bf16*)(ws + OFF_M1T);
    __bf16* m3T     = (__bf16*)(ws + OFF_M3T);
    __bf16* mlastT  = (__bf16*)(ws + OFF_MLT);
    float*  hbuf    = (float*)(ws + OFF_H);
    __bf16* actA    = (__bf16*)(ws + OFF_ACTA);
    __bf16* actB    = (__bf16*)(ws + OFF_ACTB);

    const int B = 256;
    const int waves_per_block = B / 32;

    // 0) zero accumulators (deterministic per call)
    k_zero_f32<<<((int)SMALL_FLOATS + B - 1) / B, B, 0, stream>>>(sm, (int)SMALL_FLOATS);

    // 1) prep: x -> bf16 (padded), weights -> tanh(w/2) transposed bf16 + diagsig
    {
        long long tot = (long long)M_ROWS * KP0;
        k_convert_x<<<(unsigned)((tot + B - 1) / B), B, 0, stream>>>(x, xbf);
    }
    k_prep_m0   <<<(H_DIM * KP0   + B - 1) / B, B, 0, stream>>>(w0, m0T, diag1);
    k_prep_sq   <<<(H_DIM * H_DIM + B - 1) / B, B, 0, stream>>>(w1, m1T, diag2);
    k_prep_sq   <<<(H_DIM * H_DIM + B - 1) / B, B, 0, stream>>>(w3, m3T, diag3);
    k_prep_mlast<<<(16 * H_DIM    + B - 1) / B, B, 0, stream>>>(wlast, mlastT);

    const int tiles_h   = (M_ROWS / (MT * 16)) * (H_DIM / (NT * 16));  // 1024*16
    const int blocks_h  = tiles_h / waves_per_block;                   // 2048
    const int blocks_bn = (int)(((long long)M_ROWS * H_DIM + B - 1) / B);

    // 2) layer 1: h1 = SQ2PI*(x@m0 + th0)/sqrt(diag1); x1 = tanh(BN(h1))
    k_gemm_ebp<<<blocks_h, B, 0, stream>>>(xbf, m0T, th0, diag1, hbuf, csum0, csq0,
                                           H_DIM, KP0, KP0);
    k_bn_tanh <<<blocks_bn, B, 0, stream>>>(hbuf, csum0, csq0, actA);

    // 3) layer 2: x2 = ebp(x1, m1, th1)
    k_gemm_ebp<<<blocks_h, B, 0, stream>>>(actA, m1T, th1, diag2, hbuf, csum1, csq1,
                                           H_DIM, H_DIM, H_DIM);
    k_bn_tanh <<<blocks_bn, B, 0, stream>>>(hbuf, csum1, csq1, actB);

    // 4) layer 3: x4 = ebp(x2, m3, th1)   (th1 reused, faithful to reference)
    k_gemm_ebp<<<blocks_h, B, 0, stream>>>(actB, m3T, th1, diag3, hbuf, csum2, csq2,
                                           H_DIM, H_DIM, H_DIM);
    k_bn_tanh <<<blocks_bn, B, 0, stream>>>(hbuf, csum2, csq2, actA);

    // 5) head: hlastbar = x4 @ mlast + thlast   -> d_out[0 : M*10]
    k_gemm_final<<<(M_ROWS / (MT * 16)) / waves_per_block, B, 0, stream>>>(actA, mlastT,
                                                                           thlast, out);

    // 6) log-softmax + loss + accuracy
    k_softmax_loss<<<M_ROWS / B, B, 0, stream>>>(out, target,
                                                 out + (size_t)M_ROWS * N_OUT, accums);
    k_finalize<<<1, 1, 0, stream>>>(accums, out + 2 * (size_t)M_ROWS * N_OUT);
}
